// multihead_self_attention_39393440039278
// MI455X (gfx1250) — matrix operand order repaired and checked by reading, not compiled, not run on hardware
//
#include <hip/hip_runtime.h>

typedef __bf16 bf16;
typedef __attribute__((ext_vector_type(16))) __bf16 v16bf;
typedef __attribute__((ext_vector_type(8)))  __bf16 v8bf;
typedef __attribute__((ext_vector_type(8)))  float  v8f;

#define B_  2
#define S_  2048
#define D_  1024
#define H_  16
#define DK_ 64

__device__ inline v16bf cat16(v8bf lo, v8bf hi) {
    return __builtin_shufflevector(lo, hi, 0,1,2,3,4,5,6,7,8,9,10,11,12,13,14,15);
}

// ---------------------------------------------------------------- convert ---
__global__ void f32_to_bf16_kernel(const float* __restrict__ src,
                                   bf16* __restrict__ dst, int n) {
    int i = blockIdx.x * blockDim.x + threadIdx.x;
    if (i < n) dst[i] = (bf16)src[i];
}

// ------------------------------------------------------------------- GEMM ---
// out[m,n] = sum_k A[m,k] * W[n,k]   (A: [M=4096,K=1024], W: [N=1024,K=1024])
// Each wave: 16x64 output tile, 4 accumulators, A-fragment reused 4x.
// Manual double-buffering: loads for k-step i+1 are issued before the 4 WMMAs
// of k-step i so the scheduler can overlap VMEM latency with matrix math.
// mode 0: bf16 row-major  |  mode 1: bf16 transposed (V)  |  mode 2: f32
__global__ __launch_bounds__(128)
void gemm_bf16_kernel(const bf16* __restrict__ A, const bf16* __restrict__ W,
                      bf16* __restrict__ outb, float* __restrict__ outf, int mode) {
    const int lane = threadIdx.x & 31;
    const int wave = threadIdx.x >> 5;
    const int l16  = lane & 15;
    const int half = lane >> 4;
    const int m0 = (blockIdx.x * 4 + wave) * 16;
    const int n0 = blockIdx.y * 64;

    const bf16* aRow = A + (long)(m0 + l16) * D_;   // A-fragment: lane holds row M
    const bf16* wRow = W + (long)(n0 + l16) * D_;   // B-fragments: lane holds col N

    auto loadA = [&](int kk) -> v16bf {
        v8bf lo = *(const v8bf*)(aRow + kk + half * 16);       // K = half*16 + 0..7
        v8bf hi = *(const v8bf*)(aRow + kk + half * 16 + 8);   // K = half*16 + 8..15
        return cat16(lo, hi);
    };
    auto loadB = [&](int kk, int j) -> v16bf {
        const bf16* wr = wRow + (long)(j * 16) * D_ + kk + half * 16;
        v8bf lo = *(const v8bf*)(wr);                          // K = half*16 + 0..7
        v8bf hi = *(const v8bf*)(wr + 8);                      // K = half*16 + 8..15
        return cat16(lo, hi);
    };

    v8f acc[4] = {};
    v16bf a  = loadA(0);
    v16bf b0 = loadB(0, 0), b1 = loadB(0, 1), b2 = loadB(0, 2), b3 = loadB(0, 3);

#pragma unroll 4
    for (int kk = 0; kk < D_; kk += 32) {
        v16bf an = {}, bn0 = {}, bn1 = {}, bn2 = {}, bn3 = {};
        const int kn = kk + 32;
        if (kn < D_) {                       // prefetch next k-step (clause of 10 b128)
            an  = loadA(kn);
            bn0 = loadB(kn, 0);
            bn1 = loadB(kn, 1);
            bn2 = loadB(kn, 2);
            bn3 = loadB(kn, 3);
        }
        acc[0] = __builtin_amdgcn_wmma_f32_16x16x32_bf16(false, a, false, b0,
                                                         (short)0, acc[0], false, false);
        acc[1] = __builtin_amdgcn_wmma_f32_16x16x32_bf16(false, a, false, b1,
                                                         (short)0, acc[1], false, false);
        acc[2] = __builtin_amdgcn_wmma_f32_16x16x32_bf16(false, a, false, b2,
                                                         (short)0, acc[2], false, false);
        acc[3] = __builtin_amdgcn_wmma_f32_16x16x32_bf16(false, a, false, b3,
                                                         (short)0, acc[3], false, false);
        a = an; b0 = bn0; b1 = bn1; b2 = bn2; b3 = bn3;
    }

#pragma unroll
    for (int j = 0; j < 4; ++j) {
        const int n = n0 + j * 16 + l16;
#pragma unroll
        for (int r = 0; r < 8; ++r) {
            const int m = m0 + r + 8 * half;
            const float v = acc[j][r];
            if (mode == 0) {
                outb[(long)m * D_ + n] = (bf16)v;
            } else if (mode == 1) {
                const int bb = m >> 11;           // m = b*2048 + s
                const int ss = m & (S_ - 1);
                outb[((long)(bb * D_ + n)) * S_ + ss] = (bf16)v;
            } else {
                outf[(long)m * D_ + n] = v;
            }
        }
    }
}

// ------------------------------------------------------------------- RoPE ---
__global__ void rope_kernel(bf16* __restrict__ qb, bf16* __restrict__ kb,
                            const int* __restrict__ tp) {
    const int idx = blockIdx.x * blockDim.x + threadIdx.x;   // [B*S*H*32)
    if (idx >= B_ * S_ * H_ * (DK_ / 2)) return;
    const int p = idx & 31;
    const int h = (idx >> 5) & (H_ - 1);
    const int s = (idx >> 9) & (S_ - 1);
    const int b = idx >> 20;

    const float inv = __expf(-((float)(2 * p) / (float)DK_) * __logf(10000.0f));
    const float ang = (float)tp[s] * inv;
    const float cs = __cosf(ang);
    const float sn = __sinf(ang);

    const long base = ((long)(b * S_ + s)) * D_ + h * DK_ + 2 * p;
    float q0 = (float)qb[base], q1 = (float)qb[base + 1];
    qb[base]     = (bf16)(q0 * cs - q1 * sn);
    qb[base + 1] = (bf16)(q0 * sn + q1 * cs);
    float k0 = (float)kb[base], k1 = (float)kb[base + 1];
    kb[base]     = (bf16)(k0 * cs - k1 * sn);
    kb[base + 1] = (bf16)(k0 * sn + k1 * cs);
}

// -------------------------------------------------------- flash attention ---
// One wave per (b, h, 16-row tile). Online softmax over 32-column chunks.
__global__ __launch_bounds__(32)
void attn_kernel(const bf16* __restrict__ qb, const bf16* __restrict__ kb,
                 const bf16* __restrict__ vT, bf16* __restrict__ ob) {
    __shared__ alignas(16) bf16 lp[16][32];

    const int lane = threadIdx.x;
    const int l16  = lane & 15;
    const int half = lane >> 4;
    const int it = blockIdx.x, h = blockIdx.y, b = blockIdx.z;
    const int s0 = it * 16;

    // Q A-fragments: two 32-wide k-steps covering DK=64
    const bf16* qRow = qb + ((long)(b * S_ + s0 + l16)) * D_ + h * DK_;
    v16bf aq[2];
#pragma unroll
    for (int t = 0; t < 2; ++t) {
        v8bf lo = *(const v8bf*)(qRow + 32 * t + half * 16);
        v8bf hi = *(const v8bf*)(qRow + 32 * t + half * 16 + 8);
        aq[t] = cat16(lo, hi);
    }

    float mi[8], li[8];
    v8f acc[4] = {};
#pragma unroll
    for (int r = 0; r < 8; ++r) { mi[r] = -1e30f; li[r] = 0.0f; }

    const int nchunks = (s0 + 47) >> 5;
    for (int cb = 0; cb < nchunks; ++cb) {
        const int n0 = cb * 32;

        // scores: two 16-col sub-tiles, K=64 in two WMMA steps each
        v8f sc[2] = {};
#pragma unroll
        for (int nt = 0; nt < 2; ++nt) {
            const bf16* kRow = kb + ((long)(b * S_ + n0 + nt * 16 + l16)) * D_ + h * DK_;
#pragma unroll
            for (int t = 0; t < 2; ++t) {
                v8bf lo = *(const v8bf*)(kRow + 32 * t + half * 16);
                v8bf hi = *(const v8bf*)(kRow + 32 * t + half * 16 + 8);
                v16bf bk = cat16(lo, hi);
                sc[nt] = __builtin_amdgcn_wmma_f32_16x16x32_bf16(false, aq[t], false, bk,
                                                                 (short)0, sc[nt], false, false);
            }
        }

        // online softmax, row-wise across the 16-lane halves
#pragma unroll
        for (int r = 0; r < 8; ++r) {
            const int m = s0 + r + 8 * half;
            float v0 = sc[0][r] * 0.125f;                 // 1/sqrt(64)
            float v1 = sc[1][r] * 0.125f;
            if (n0 + l16 > m)      v0 = -1e30f;           // causal mask
            if (n0 + 16 + l16 > m) v1 = -1e30f;

            float cm = fmaxf(v0, v1);
#pragma unroll
            for (int msk = 1; msk < 16; msk <<= 1) cm = fmaxf(cm, __shfl_xor(cm, msk, 32));
            const float mnew  = fmaxf(mi[r], cm);
            const float alpha = __expf(mi[r] - mnew);
            const float p0 = __expf(v0 - mnew);
            const float p1 = __expf(v1 - mnew);
            float rs = p0 + p1;
#pragma unroll
            for (int msk = 1; msk < 16; msk <<= 1) rs += __shfl_xor(rs, msk, 32);
            li[r] = li[r] * alpha + rs;
            mi[r] = mnew;
#pragma unroll
            for (int t = 0; t < 4; ++t) acc[t][r] *= alpha;

            const int row = r + 8 * half;                 // C-layout -> LDS row-major
            lp[row][l16]      = (bf16)p0;
            lp[row][16 + l16] = (bf16)p1;
        }
        __syncthreads();

        // reload P as A-fragment (K-dim = sequence chunk of 32)
        v8bf plo = *(const v8bf*)(&lp[l16][half * 16]);
        v8bf phi = *(const v8bf*)(&lp[l16][half * 16 + 8]);
        v16bf pa = cat16(plo, phi);

#pragma unroll
        for (int t = 0; t < 4; ++t) {                      // 4 dv tiles of 16
            const bf16* vRow = vT + ((long)(b * D_ + h * DK_ + t * 16 + l16)) * S_
                                  + n0 + half * 16;
            v8bf lo = *(const v8bf*)(vRow);
            v8bf hi = *(const v8bf*)(vRow + 8);
            v16bf bv = cat16(lo, hi);
            acc[t] = __builtin_amdgcn_wmma_f32_16x16x32_bf16(false, pa, false, bv,
                                                             (short)0, acc[t], false, false);
        }
        __syncthreads();
    }

    // epilogue: o = acc / l, bf16 row-major
#pragma unroll
    for (int t = 0; t < 4; ++t) {
#pragma unroll
        for (int r = 0; r < 8; ++r) {
            const int m  = s0 + r + 8 * half;
            const int dv = h * DK_ + t * 16 + l16;
            ob[((long)(b * S_ + m)) * D_ + dv] = (bf16)(acc[t][r] / li[r]);
        }
    }
}

// ----------------------------------------------------------------- launch ---
extern "C" void kernel_launch(void* const* d_in, const int* in_sizes, int n_in,
                              void* d_out, int out_size, void* d_ws, size_t ws_size,
                              hipStream_t stream) {
    const float* x  = (const float*)d_in[0];
    const float* Qw = (const float*)d_in[1];
    const float* Kw = (const float*)d_in[2];
    const float* Vw = (const float*)d_in[3];
    const float* Ow = (const float*)d_in[4];
    const int*   tp = (const int*)d_in[5];

    char* ws = (char*)d_ws;
    bf16* xb = (bf16*)(ws);                        //  8 MB: x bf16
    bf16* wq = (bf16*)(ws + ( 8l << 20));          //  2 MB each
    bf16* wk = (bf16*)(ws + (10l << 20));
    bf16* wv = (bf16*)(ws + (12l << 20));
    bf16* wo = (bf16*)(ws + (14l << 20));
    bf16* qb = (bf16*)(ws + (16l << 20));          //  8 MB
    bf16* kb = (bf16*)(ws + (24l << 20));          //  8 MB
    bf16* vT = (bf16*)(ws + (32l << 20));          //  8 MB (transposed)
    bf16* ob = (bf16*)(ws + (40l << 20));          //  8 MB

    const int nx = B_ * S_ * D_;
    const int nw = D_ * D_;
    f32_to_bf16_kernel<<<(nx + 255) / 256, 256, 0, stream>>>(x,  xb, nx);
    f32_to_bf16_kernel<<<(nw + 255) / 256, 256, 0, stream>>>(Qw, wq, nw);
    f32_to_bf16_kernel<<<(nw + 255) / 256, 256, 0, stream>>>(Kw, wk, nw);
    f32_to_bf16_kernel<<<(nw + 255) / 256, 256, 0, stream>>>(Vw, wv, nw);
    f32_to_bf16_kernel<<<(nw + 255) / 256, 256, 0, stream>>>(Ow, wo, nw);

    // 64x64 block tiles: 4 waves/block, each wave does a 16x64 tile
    dim3 gg(B_ * S_ / 64, D_ / 64);                // (64, 16) blocks
    gemm_bf16_kernel<<<gg, 128, 0, stream>>>(xb, wq, qb, nullptr, 0);
    gemm_bf16_kernel<<<gg, 128, 0, stream>>>(xb, wk, kb, nullptr, 0);
    gemm_bf16_kernel<<<gg, 128, 0, stream>>>(xb, wv, vT, nullptr, 1);

    const int np = B_ * S_ * H_ * (DK_ / 2);
    rope_kernel<<<(np + 255) / 256, 256, 0, stream>>>(qb, kb, tp);

    attn_kernel<<<dim3(S_ / 16, H_, B_), 32, 0, stream>>>(qb, kb, vT, ob);

    gemm_bf16_kernel<<<gg, 128, 0, stream>>>(ob, wo, nullptr, (float*)d_out, 2);
}
